// NeighSim_84267258347859
// MI455X (gfx1250) — matrix operand order repaired and checked by reading, not compiled, not run on hardware
//
#include <hip/hip_runtime.h>

// NeighSim on MI455X (gfx1250, wave32).
// Dims fixed per reference: B=8, C=128, H=W=128, RADIUS=1, NB_ITER=2.
#define Bn 8
#define Cn 128
#define Hn 128
#define Wn 128
#define PITCH 136          // f16 per staged pixel (pad: 272B row, 16B aligned)
#define PADV 1e-4f
#define SFACT 50.0f

typedef __attribute__((ext_vector_type(16))) _Float16 v16h;
typedef __attribute__((ext_vector_type(8)))  _Float16 v8h;
typedef __attribute__((ext_vector_type(8)))  float    v8f;

// ---------------- pass 0: per-pixel L2 channel normalization ----------------
__global__ __launch_bounds__(128) void normalize_k(const float* __restrict__ in,
                                                   float* __restrict__ out) {
  const int w = threadIdx.x;
  const int h = blockIdx.x % Hn;
  const int b = blockIdx.x / Hn;
  const int base = ((b * Cn) * Hn + h) * Wn + w;     // channel stride = Hn*Wn
  float s = 0.f;
  for (int c = 0; c < Cn; ++c) { float v = in[base + c * Hn * Wn]; s += v * v; }
  const float inv = 1.0f / sqrtf(s);
  for (int c = 0; c < Cn; ++c) out[base + c * Hn * Wn] = in[base + c * Hn * Wn] * inv;
}

// ---------------- one NeighSim iteration: xin -> xout ----------------
// One workgroup (8 wave32) per (b, h) row of 128 pixels.
__global__ __launch_bounds__(256) void neigh_iter(const float* __restrict__ xin,
                                                  float* __restrict__ xout) {
  __shared__ __align__(16) _Float16 xh[3 * Wn * PITCH];  // 104448 B, rows h-1,h,h+1 (clamped)
  __shared__ float gram[24][16][16];                     // 24576 B: 8 col tiles x 3 row pairs
  __shared__ float nrm[3][Wn];                           // per-pixel norms (from f16 data)
  __shared__ float bdot[3][2][8];                        // cross-tile band entries
  __shared__ float wgt[8][Wn];                           // softmax weight * mask

  const int tid = threadIdx.x;
  const int h = blockIdx.x % Hn;
  const int b = blockIdx.x / Hn;

  // ---- stage: global f32 [c][w] -> LDS f16 [pixel][c] (coalesced in w) ----
  for (int idx = tid; idx < 3 * Cn * Wn; idx += 256) {
    const int r = idx / (Cn * Wn);
    const int c = (idx / Wn) % Cn;
    const int w = idx % Wn;
    const int hr = min(max(h - 1 + r, 0), Hn - 1);       // replicate clamp
    xh[(r * Wn + w) * PITCH + c] = (_Float16)xin[((b * Cn + c) * Hn + hr) * Wn + w];
  }
  __syncthreads();

  // ---- norms of all 3x128 staged pixels ----
  for (int j = tid; j < 3 * Wn; j += 256) {
    const _Float16* p = &xh[j * PITCH];
    float s = 0.f;
    for (int c = 0; c < Cn; ++c) { float v = (float)p[c]; s += v * v; }
    nrm[j / Wn][j % Wn] = sqrtf(s);
  }

  // ---- 42 cross-tile band dots the diagonal Gram blocks miss ----
  if (tid < 42) {
    const int r = tid / 14, id2 = tid % 14;
    int q, k, side, t;
    if (id2 < 7) { t = id2 + 1; q = 16 * t;      k = q - 1; side = 0; }
    else         { t = id2 - 7; q = 16 * t + 15; k = q + 1; side = 1; }
    const _Float16* qa = &xh[(1 * Wn + q) * PITCH];
    const _Float16* ka = &xh[(r * Wn + k) * PITCH];
    float s = 0.f;
    for (int c = 0; c < Cn; ++c) s += (float)qa[c] * (float)ka[c];
    bdot[r][side][t] = s;
  }

  // ---- WMMA: 24 diagonal Gram blocks, D = Q^T K accumulated over C=128 ----
  // Wave wv owns column tile t = wv; compile-time loop over the 3 row pairs
  // keeps trip counts uniform (EXEC all-1s around every WMMA, per ISA 7.12).
  {
    const int wv = tid >> 5, lane = tid & 31;
    const int m = lane & 15, hi = lane >> 4;             // lane half per ISA layouts
    const _Float16* qb = &xh[(1 * Wn + 16 * wv + m) * PITCH];    // A: M = query pixel
    #pragma unroll
    for (int r = 0; r < 3; ++r) {
      const _Float16* kb = &xh[(r * Wn + 16 * wv + m) * PITCH];  // B: N = key pixel
      v8f acc = {};
      #pragma unroll
      for (int kc = 0; kc < Cn; kc += 32) {
        // A (16x32 f16): lanes 0-15 hold K = kc+{0..7,16..23}, lanes 16-31 K += 8
        v8h a0 = *(const v8h*)(qb + kc + hi * 8);
        v8h a1 = *(const v8h*)(qb + kc + 16 + hi * 8);
        // B (32x16 f16): lanes 0-15 hold K = kc+{0..7,16..23}, lanes 16-31 K += 8
        v8h b0 = *(const v8h*)(kb + kc + hi * 8);
        v8h b1 = *(const v8h*)(kb + kc + 16 + hi * 8);
        v16h A  = __builtin_shufflevector(a0, a1, 0,1,2,3,4,5,6,7,8,9,10,11,12,13,14,15);
        v16h Bm = __builtin_shufflevector(b0, b1, 0,1,2,3,4,5,6,7,8,9,10,11,12,13,14,15);
        acc = __builtin_amdgcn_wmma_f32_16x16x32_f16(false, A, false, Bm,
                                                     (short)0, acc, false, false);
      }
      // D layout: lanes 0-15 -> N=lane, M=g; lanes 16-31 -> N=lane-16, M=g+8
      #pragma unroll
      for (int g = 0; g < 8; ++g) gram[r * 8 + wv][g + hi * 8][m] = acc[g];
    }
  }
  __syncthreads();

  // ---- sims + softmax(50*sim) -> aggregation weights (incl. feat mask) ----
  if (tid < Wn) {
    const int w = tid, t = w >> 4, m = w & 15;
    const float nq = nrm[1][w];
    float s[8], mk[8];
    int di = 0;
    for (int dh = -1; dh <= 1; ++dh)
      for (int dw = -1; dw <= 1; ++dw) {
        if (dh == 0 && dw == 0) continue;
        const int r = dh + 1;
        float mask = ((unsigned)(h + dh) < (unsigned)Hn) ? 1.f : PADV;
        int wp = w + dw;
        mask *= ((unsigned)wp < (unsigned)Wn) ? 1.f : PADV;   // corners: PAD*PAD
        wp = min(max(wp, 0), Wn - 1);
        float dot;
        if ((wp >> 4) == t) dot = gram[r * 8 + t][m][wp & 15];
        else if (wp < w)    dot = bdot[r][0][t];
        else                dot = bdot[r][1][t];
        s[di]  = mask * dot / (nrm[r][wp] * nq);
        mk[di] = mask;
        ++di;
      }
    float mx = s[0];
    for (int d = 1; d < 8; ++d) mx = fmaxf(mx, s[d]);
    float e[8], sum = 0.f;
    for (int d = 0; d < 8; ++d) { e[d] = __expf(SFACT * (s[d] - mx)); sum += e[d]; }
    const float isum = 1.0f / sum;
    for (int d = 0; d < 8; ++d) wgt[d][w] = e[d] * isum * mk[d];
  }
  __syncthreads();

  // ---- aggregation: 2 threads per pixel, 64 channels each, b128 LDS reads ----
  {
    const int w  = tid & (Wn - 1);
    const int c0 = (tid >> 7) * (Cn / 2);
    float wl[8]; int off[8];
    int di = 0;
    for (int dh = -1; dh <= 1; ++dh)
      for (int dw = -1; dw <= 1; ++dw) {
        if (dh == 0 && dw == 0) continue;
        const int wp = min(max(w + dw, 0), Wn - 1);
        off[di] = ((dh + 1) * Wn + wp) * PITCH;
        wl[di]  = wgt[di][w];
        ++di;
      }
    for (int c = c0; c < c0 + Cn / 2; c += 8) {          // 8 channels per step
      float a[8] = {0.f, 0.f, 0.f, 0.f, 0.f, 0.f, 0.f, 0.f};
      #pragma unroll
      for (int d = 0; d < 8; ++d) {
        v8h v = *(const v8h*)(&xh[off[d] + c]);          // 16B-aligned ds_load_b128
        #pragma unroll
        for (int j = 0; j < 8; ++j) a[j] += wl[d] * (float)v[j];
      }
      #pragma unroll
      for (int j = 0; j < 8; ++j)
        xout[((b * Cn + (c + j)) * Hn + h) * Wn + w] = a[j];   // coalesced in w
    }
  }
}

extern "C" void kernel_launch(void* const* d_in, const int* in_sizes, int n_in,
                              void* d_out, int out_size, void* d_ws, size_t ws_size,
                              hipStream_t stream) {
  (void)in_sizes; (void)n_in; (void)out_size; (void)ws_size;
  const float* x  = (const float*)d_in[0];
  float* out = (float*)d_out;
  float* ws  = (float*)d_ws;          // needs >= 64 MB (one field) for ping-pong

  normalize_k<<<Bn * Hn, 128, 0, stream>>>(x, out);   // normalized field -> out
  neigh_iter <<<Bn * Hn, 256, 0, stream>>>(out, ws);  // iteration 1
  neigh_iter <<<Bn * Hn, 256, 0, stream>>>(ws, out);  // iteration 2 -> d_out
}
